// Attention_56118042689620
// MI455X (gfx1250) — matrix operand order repaired and checked by reading, not compiled, not run on hardware
//
#include <hip/hip_runtime.h>
#include <hip/hip_bf16.h>

#define DIM   2048
#define NH    16
#define HD    128
#define BATCH 2
#define SEQ   2048

typedef __bf16 bf16_t;
typedef __attribute__((ext_vector_type(16))) __bf16 v16bf;
typedef __attribute__((ext_vector_type(2)))  __bf16 v2bf;
typedef __attribute__((ext_vector_type(8)))  float  v8f;

union Frag16 {
  v16bf v;
  uint4 u[2];
};

// Native f32 -> bf16 (lowered to v_cvt_*bf16* on gfx1250)
__device__ __forceinline__ bf16_t f2bf(float f) { return (bf16_t)f; }

// Pack two f32 into a DWORD of 2x bf16 (k, k+1)
__device__ __forceinline__ unsigned pkbf(float a, float b) {
  union { v2bf v; unsigned u; } p;
  p.v[0] = (bf16_t)a;
  p.v[1] = (bf16_t)b;
  return p.u;
}

__device__ __forceinline__ v8f v8f_zero() {
  v8f z;
#pragma unroll
  for (int i = 0; i < 8; ++i) z[i] = 0.0f;
  return z;
}

// ---------------------------------------------------------------------------
// Kernel 1: fused QKV projection (x @ [wq|wk|wv]) + RoPE, bf16 WMMA.
//   grid = (6144/128, 4096/128) = (48, 32), block = 256 (8 waves).
//   Block tile 128x128, K-step 32. Wave tile 64x32 = 4x2 WMMA 16x16 tiles.
//   q,k stored [b][h][s][128] (roped); v stored transposed [b][h][d][s].
// ---------------------------------------------------------------------------
__global__ __launch_bounds__(256) void qkv_rope_gemm(
    const float* __restrict__ x,
    const float* __restrict__ wq, const float* __restrict__ wk,
    const float* __restrict__ wv,
    const float* __restrict__ fcos, const float* __restrict__ fsin,
    bf16_t* __restrict__ qout, bf16_t* __restrict__ kout,
    bf16_t* __restrict__ vT) {
  __shared__ bf16_t As[128][32];   // [m][k]
  __shared__ bf16_t Bs[128][32];   // transposed: [n][k]

  const int tid  = threadIdx.x;
  const int lane = tid & 31;
  const int wave = tid >> 5;
  const int half = lane >> 4;      // 0: rows 0-7 / K-lo, 1: rows 8-15 / K-hi
  const int col  = lane & 15;

  const int m0    = blockIdx.y * 128;           // row in [0, 4096)
  const int n0g   = blockIdx.x * 128;           // col in [0, 6144)
  const int which = n0g >> 11;                  // 0=q, 1=k, 2=v
  const int n0    = n0g & (DIM - 1);
  const float* wptr = (which == 0) ? wq : ((which == 1) ? wk : wv);

  const int wm = wave >> 2;                     // 0..1
  const int wn = wave & 3;                      // 0..3

  v8f acc[4][2];
#pragma unroll
  for (int i = 0; i < 4; ++i)
#pragma unroll
    for (int j = 0; j < 2; ++j) acc[i][j] = v8f_zero();

  // global->LDS staging assignments
  const int arow  = tid >> 1;            // 0..127
  const int acol  = (tid & 1) << 4;      // 0 / 16
  const int brow2 = (tid >> 4) << 1;     // even k: 0..30
  const int bncol = (tid & 15) << 3;     // n: 0..120

  for (int k0 = 0; k0 < DIM; k0 += 32) {
    const float4* ap  = (const float4*)(x + (size_t)(m0 + arow) * DIM + k0 + acol);
    const float4* bp0 = (const float4*)(wptr + (size_t)(k0 + brow2) * DIM + n0 + bncol);
    const float4* bp1 = (const float4*)(wptr + (size_t)(k0 + brow2 + 1) * DIM + n0 + bncol);

    // A: 16 consecutive k, converted in regs, two b128 LDS stores
    Frag16 apk;
#pragma unroll
    for (int i = 0; i < 4; ++i) {
      float4 va = ap[i];
      apk.v[i * 4 + 0] = f2bf(va.x);
      apk.v[i * 4 + 1] = f2bf(va.y);
      apk.v[i * 4 + 2] = f2bf(va.z);
      apk.v[i * 4 + 3] = f2bf(va.w);
    }
    *(uint4*)&As[arow][acol]     = apk.u[0];
    *(uint4*)&As[arow][acol + 8] = apk.u[1];

    // B: 2 k-rows x 8 n, packed (k,k+1) DWORD stores into transposed tile
    float4 r0a = bp0[0], r0b = bp0[1];
    float4 r1a = bp1[0], r1b = bp1[1];
    *(unsigned*)&Bs[bncol + 0][brow2] = pkbf(r0a.x, r1a.x);
    *(unsigned*)&Bs[bncol + 1][brow2] = pkbf(r0a.y, r1a.y);
    *(unsigned*)&Bs[bncol + 2][brow2] = pkbf(r0a.z, r1a.z);
    *(unsigned*)&Bs[bncol + 3][brow2] = pkbf(r0a.w, r1a.w);
    *(unsigned*)&Bs[bncol + 4][brow2] = pkbf(r0b.x, r1b.x);
    *(unsigned*)&Bs[bncol + 5][brow2] = pkbf(r0b.y, r1b.y);
    *(unsigned*)&Bs[bncol + 6][brow2] = pkbf(r0b.z, r1b.z);
    *(unsigned*)&Bs[bncol + 7][brow2] = pkbf(r0b.w, r1b.w);

    if (k0 + 32 < DIM) __builtin_prefetch((const char*)ap + 128, 0, 0);
    __syncthreads();

    // A fragment: lane<16 -> K {0..7,16..23}, lane>=16 -> K {8..15,24..31}
    // B fragment: lane<16 -> K {0..7,16..23}, lane>=16 -> K {8..15,24..31}
    Frag16 af[4], bfr[2];
    const int koffA = half * 8;
    const int koffB = half * 8;
#pragma unroll
    for (int i = 0; i < 4; ++i) {
      const int r = wm * 64 + i * 16 + col;
      af[i].u[0] = *(const uint4*)&As[r][koffA];
      af[i].u[1] = *(const uint4*)&As[r][16 + koffA];
    }
#pragma unroll
    for (int j = 0; j < 2; ++j) {
      const int nn = wn * 32 + j * 16 + col;
      bfr[j].u[0] = *(const uint4*)&Bs[nn][koffB];
      bfr[j].u[1] = *(const uint4*)&Bs[nn][koffB + 16];
    }
#pragma unroll
    for (int i = 0; i < 4; ++i)
#pragma unroll
      for (int j = 0; j < 2; ++j)
        acc[i][j] = __builtin_amdgcn_wmma_f32_16x16x32_bf16(
            false, af[i].v, false, bfr[j].v, (short)0, acc[i][j], false, false);
    __syncthreads();
  }

  // Epilogue: RoPE (q,k) + scatter to attention layouts.
  // C layout: lane = column n, vgpr r = row (half*8 + r).
#pragma unroll
  for (int i = 0; i < 4; ++i) {
    const int mbase = m0 + wm * 64 + i * 16 + half * 8;
#pragma unroll
    for (int j = 0; j < 2; ++j) {
      const int n = n0 + wn * 32 + j * 16 + col;   // 0..2047 within weight
      const int h = n >> 7;
      const int d = n & (HD - 1);
      if (which < 2) {
        bf16_t* outp = (which == 0) ? qout : kout;
        const int iidx = d >> 1;
#pragma unroll
        for (int r = 0; r < 8; ++r) {
          const int m  = mbase + r;
          const int bb = m >> 11;
          const int s  = m & (SEQ - 1);
          const float vcur  = acc[i][j][r];
          const float vpart = __shfl_xor(vcur, 1, 32);  // pairs n <-> n^1
          const float c  = fcos[s * (HD / 2) + iidx];
          const float sn = fsin[s * (HD / 2) + iidx];
          const float o  = (d & 1) ? (vpart * sn + vcur * c)
                                   : (vcur * c - vpart * sn);
          outp[(((size_t)bb * NH + h) * SEQ + s) * HD + d] = f2bf(o);
        }
      } else {
#pragma unroll
        for (int r = 0; r < 8; ++r) {
          const int m  = mbase + r;
          const int bb = m >> 11;
          const int s  = m & (SEQ - 1);
          vT[(((size_t)bb * NH + h) * HD + d) * SEQ + s] = f2bf(acc[i][j][r]);
        }
      }
    }
  }
}

// ---------------------------------------------------------------------------
// Kernel 2: flash attention, causal, bf16 WMMA, online softmax.
//   grid = (SEQ/128, BATCH*NH) = (16, 32), block = 256 (8 waves).
//   Each wave owns 16 query rows; computes S^T tiles (M=key, N=query) so the
//   softmax is lane-local and P repacks register-locally into the PV A-frag.
// ---------------------------------------------------------------------------
__global__ __launch_bounds__(256) void flash_attn(
    const bf16_t* __restrict__ q, const bf16_t* __restrict__ k,
    const bf16_t* __restrict__ vT, bf16_t* __restrict__ attn) {
  const int tid  = threadIdx.x;
  const int lane = tid & 31;
  const int wave = tid >> 5;
  const int half = lane >> 4;
  const int col  = lane & 15;

  const int bh = blockIdx.y;           // 0..31
  const int b  = bh >> 4;
  const int h  = bh & (NH - 1);
  const int q0 = blockIdx.x * 128 + wave * 16;
  const int qg = q0 + col;             // this lane's query row (S^T column)

  const size_t qkbase = (size_t)bh * SEQ * HD;   // q/k: [bh][s][128]
  const size_t vbase  = (size_t)bh * HD * SEQ;   // vT : [bh][d][s]

  // Hoist Q fragments (B-operand of S^T = K·Qᵀ): 4 chunks of 32 over d.
  Frag16 qf[4];
  const bf16_t* qrow = q + qkbase + (size_t)qg * HD;
#pragma unroll
  for (int kk = 0; kk < 4; ++kk) {
    const uint4* p = (const uint4*)(qrow + kk * 32 + half * 8);
    qf[kk].u[0] = p[0];
    qf[kk].u[1] = p[2];
  }

  v8f o[8];
#pragma unroll
  for (int g = 0; g < 8; ++g) o[g] = v8f_zero();
  float mrun = -3.0e38f, lrun = 0.0f;
  const float scale = 0.08838834764831845f;   // 1/sqrt(128)

  const int kend = q0 + 15;                    // causal upper bound
  for (int k0 = 0; k0 <= kend; k0 += 32) {
    // --- S^T tiles: keys [k0,k0+16) and [k0+16,k0+32) vs 16 queries -------
    v8f st[2];
    st[0] = v8f_zero(); st[1] = v8f_zero();
#pragma unroll
    for (int t = 0; t < 2; ++t) {
#pragma unroll
      for (int kk = 0; kk < 4; ++kk) {
        const bf16_t* krow =
            k + qkbase + (size_t)(k0 + t * 16 + col) * HD + kk * 32;
        Frag16 kf;
        kf.u[0] = *(const uint4*)(krow + half * 8);
        kf.u[1] = *(const uint4*)(krow + 16 + half * 8);
        st[t] = __builtin_amdgcn_wmma_f32_16x16x32_bf16(
            false, kf.v, false, qf[kk].v, (short)0, st[t], false, false);
      }
    }

    // --- online softmax (lane = one query column) -------------------------
    float p[16];
    float tmax = -3.0e38f;
#pragma unroll
    for (int t = 0; t < 2; ++t)
#pragma unroll
      for (int r = 0; r < 8; ++r) {
        const int key = k0 + t * 16 + half * 8 + r;
        float sv = st[t][r] * scale;
        if (key > qg) sv = -3.0e38f;           // causal mask
        p[t * 8 + r] = sv;
        tmax = fmaxf(tmax, sv);
      }
    tmax = fmaxf(tmax, __shfl_xor(tmax, 16, 32));  // merge key halves
    const float mnew  = fmaxf(mrun, tmax);
    const float alpha = __expf(mrun - mnew);
    float lsum = 0.0f;
    Frag16 pf;   // register-local repack: A-operand of P·V (16x32)
#pragma unroll
    for (int e = 0; e < 16; ++e) {
      const float pv = __expf(p[e] - mnew);
      lsum += pv;
      pf.v[e] = f2bf(pv);
    }
    lsum += __shfl_xor(lsum, 16, 32);
    lrun = lrun * alpha + lsum;
    mrun = mnew;

    // rescale O: per-row alpha broadcast from "lane == query" layout
#pragma unroll
    for (int r = 0; r < 8; ++r) {
      const float ar = __shfl(alpha, half * 8 + r, 32);
#pragma unroll
      for (int g = 0; g < 8; ++g) o[g][r] *= ar;
    }

    // --- P·V: 8 output d-groups, K=32 keys, B from transposed V ----------
#pragma unroll
    for (int g = 0; g < 8; ++g) {
      const bf16_t* vrow =
          vT + vbase + (size_t)(g * 16 + col) * SEQ + k0 + half * 8;
      Frag16 vf;
      vf.u[0] = ((const uint4*)vrow)[0];
      vf.u[1] = ((const uint4*)vrow)[2];
      o[g] = __builtin_amdgcn_wmma_f32_16x16x32_bf16(
          false, pf.v, false, vf.v, (short)0, o[g], false, false);
    }
  }

  // normalize and store bf16 to attn [b][s][h*128+d]
  float lr[8];
#pragma unroll
  for (int r = 0; r < 8; ++r)
    lr[r] = 1.0f / __shfl(lrun, half * 8 + r, 32);
  const size_t obase = (size_t)b * SEQ * DIM;
#pragma unroll
  for (int g = 0; g < 8; ++g) {
    const int d = g * 16 + col;
#pragma unroll
    for (int r = 0; r < 8; ++r) {
      const int s = q0 + half * 8 + r;
      attn[obase + (size_t)s * DIM + h * HD + d] = f2bf(o[g][r] * lr[r]);
    }
  }
}

// ---------------------------------------------------------------------------
// Kernel 3: output projection attn(bf16) @ wo(f32->bf16) -> out(f32).
//   grid = (2048/128, 4096/128) = (16, 32), block = 256.
//   A-tile staging uses gfx1250 async global->LDS DMA (no VGPR round-trip),
//   tracked by ASYNCcnt and drained with s_wait_asynccnt before the barrier.
// ---------------------------------------------------------------------------
__global__ __launch_bounds__(256) void out_proj_gemm(
    const bf16_t* __restrict__ attn, const float* __restrict__ wo,
    float* __restrict__ out) {
  __shared__ bf16_t As[128][32];
  __shared__ bf16_t Bs[128][32];

  const int tid  = threadIdx.x;
  const int lane = tid & 31;
  const int wave = tid >> 5;
  const int half = lane >> 4;
  const int col  = lane & 15;

  const int m0 = blockIdx.y * 128;
  const int n0 = blockIdx.x * 128;
  const int wm = wave >> 2;
  const int wn = wave & 3;

  v8f acc[4][2];
#pragma unroll
  for (int i = 0; i < 4; ++i)
#pragma unroll
    for (int j = 0; j < 2; ++j) acc[i][j] = v8f_zero();

  const int arow  = tid >> 1;
  const int acol  = (tid & 1) << 4;
  const int brow2 = (tid >> 4) << 1;
  const int bncol = (tid & 15) << 3;

  const unsigned lds_a = (unsigned)(size_t)&As[arow][acol];

  for (int k0 = 0; k0 < DIM; k0 += 32) {
    // A (bf16, contiguous): async DMA global -> LDS, 2 x 16B per thread
    const unsigned long long ga =
        (unsigned long long)(attn + (size_t)(m0 + arow) * DIM + k0 + acol);
    asm volatile(
        "global_load_async_to_lds_b128 %0, %1, off\n\t"
        "global_load_async_to_lds_b128 %0, %1, off offset:16"
        :: "v"(lds_a), "v"(ga) : "memory");

    // B: f32 -> bf16, 2 k-rows x 8 n, packed DWORD transposed stores
    const float4* bp0 = (const float4*)(wo + (size_t)(k0 + brow2) * DIM + n0 + bncol);
    const float4* bp1 = (const float4*)(wo + (size_t)(k0 + brow2 + 1) * DIM + n0 + bncol);
    float4 r0a = bp0[0], r0b = bp0[1];
    float4 r1a = bp1[0], r1b = bp1[1];
    *(unsigned*)&Bs[bncol + 0][brow2] = pkbf(r0a.x, r1a.x);
    *(unsigned*)&Bs[bncol + 1][brow2] = pkbf(r0a.y, r1a.y);
    *(unsigned*)&Bs[bncol + 2][brow2] = pkbf(r0a.z, r1a.z);
    *(unsigned*)&Bs[bncol + 3][brow2] = pkbf(r0a.w, r1a.w);
    *(unsigned*)&Bs[bncol + 4][brow2] = pkbf(r0b.x, r1b.x);
    *(unsigned*)&Bs[bncol + 5][brow2] = pkbf(r0b.y, r1b.y);
    *(unsigned*)&Bs[bncol + 6][brow2] = pkbf(r0b.z, r1b.z);
    *(unsigned*)&Bs[bncol + 7][brow2] = pkbf(r0b.w, r1b.w);

    asm volatile("s_wait_asynccnt 0x0" ::: "memory");
    __syncthreads();

    Frag16 af[4], bfr[2];
    const int koffA = half * 8;
    const int koffB = half * 8;
#pragma unroll
    for (int i = 0; i < 4; ++i) {
      const int r = wm * 64 + i * 16 + col;
      af[i].u[0] = *(const uint4*)&As[r][koffA];
      af[i].u[1] = *(const uint4*)&As[r][16 + koffA];
    }
#pragma unroll
    for (int j = 0; j < 2; ++j) {
      const int nn = wn * 32 + j * 16 + col;
      bfr[j].u[0] = *(const uint4*)&Bs[nn][koffB];
      bfr[j].u[1] = *(const uint4*)&Bs[nn][koffB + 16];
    }
#pragma unroll
    for (int i = 0; i < 4; ++i)
#pragma unroll
      for (int j = 0; j < 2; ++j)
        acc[i][j] = __builtin_amdgcn_wmma_f32_16x16x32_bf16(
            false, af[i].v, false, bfr[j].v, (short)0, acc[i][j], false, false);
    __syncthreads();
  }

#pragma unroll
  for (int i = 0; i < 4; ++i) {
    const int mbase = m0 + wm * 64 + i * 16 + half * 8;
#pragma unroll
    for (int j = 0; j < 2; ++j) {
      const int n = n0 + wn * 32 + j * 16 + col;
#pragma unroll
      for (int r = 0; r < 8; ++r)
        out[(size_t)(mbase + r) * DIM + n] = acc[i][j][r];
    }
  }
}

// ---------------------------------------------------------------------------
extern "C" void kernel_launch(void* const* d_in, const int* in_sizes, int n_in,
                              void* d_out, int out_size, void* d_ws,
                              size_t ws_size, hipStream_t stream) {
  const float* x    = (const float*)d_in[0];
  const float* wq   = (const float*)d_in[1];
  const float* wk   = (const float*)d_in[2];
  const float* wv   = (const float*)d_in[3];
  const float* wo   = (const float*)d_in[4];
  const float* fcos = (const float*)d_in[5];
  const float* fsin = (const float*)d_in[6];
  // d_in[7] (mask) unused: causal mask computed inline.

  const size_t elems = (size_t)BATCH * SEQ * DIM;   // 8,388,608
  bf16_t* qws = (bf16_t*)d_ws;
  bf16_t* kws = qws + elems;
  bf16_t* vws = kws + elems;
  bf16_t* aws = vws + elems;

  qkv_rope_gemm<<<dim3(48, 32), 256, 0, stream>>>(x, wq, wk, wv, fcos, fsin,
                                                  qws, kws, vws);
  flash_attn<<<dim3(16, 32), 256, 0, stream>>>(qws, kws, vws, aws);
  out_proj_gemm<<<dim3(16, 32), 256, 0, stream>>>(aws, wo, (float*)d_out);
}
